// TemporalCrossAttention_87222195847991
// MI455X (gfx1250) — matrix operand order repaired and checked by reading, not compiled, not run on hardware
//
#include <hip/hip_runtime.h>
#include <hip/hip_bf16.h>
#include <math.h>

// Problem dims (compile-time)
#define BB    4
#define TDIM  2048
#define DDIM  1024
#define NDIM  512
#define LDIM  512
#define HH    16
#define DHEAD 64

typedef __attribute__((ext_vector_type(16))) _Float16 v16h;
typedef __attribute__((ext_vector_type(8)))  _Float16 v8h;
typedef __attribute__((ext_vector_type(8)))  float    v8f;

union AFrag { v16h v; v8h h[2]; _Float16 e[16]; };
union CFrag { v8f v; float e[8]; };

// ---- WMMA fragment loaders (CDNA5 wave32 layouts, 16x16x32 f16) ----
// A (16x32, f16), p points at (row0, k0) of a row-major matrix, leading dim ld.
// lane l: M = l&15; halves h -> K = h + (l>=16 ? 16:0)
__device__ __forceinline__ v16h load_a_frag(const _Float16* p, int ld, int lane) {
  const int m = lane & 15, hi = lane >> 4;
  const _Float16* q = p + (size_t)m * ld + hi * 16;
  AFrag f;
  f.h[0] = *(const v8h*)(q);       // K = hi*16 + 0..7
  f.h[1] = *(const v8h*)(q + 8);   // K = hi*16 + 8..15
  return f.v;
}
// B (32x16, f16) loaded from a K-major ("transposed") matrix: p points at
// (n0, k0) of an [N][K] row-major matrix. lane l: N = l&15;
// halves h -> K = h + (l>=16 ? 16:0)  => one contiguous 16-half run per lane.
__device__ __forceinline__ v16h load_b_frag(const _Float16* p, int ld, int lane) {
  const int n = lane & 15, hi = lane >> 4;
  return *(const v16h*)(p + (size_t)n * ld + hi * 16);
}

// ---- LayerNorm -> f16 ----
__global__ void ln_f16_kernel(const float* __restrict__ x, const float* __restrict__ w,
                              const float* __restrict__ b, _Float16* __restrict__ out, int C) {
  __shared__ float rs[256], rs2[256];
  const int row = blockIdx.x, tid = threadIdx.x;
  const float* xr = x + (size_t)row * C;
  float s = 0.f, s2 = 0.f;
  for (int i = tid; i < C; i += 256) { float v = xr[i]; s += v; s2 += v * v; }
  rs[tid] = s; rs2[tid] = s2; __syncthreads();
  for (int st = 128; st > 0; st >>= 1) {
    if (tid < st) { rs[tid] += rs[tid + st]; rs2[tid] += rs2[tid + st]; }
    __syncthreads();
  }
  const float mu  = rs[0] / (float)C;
  const float var = rs2[0] / (float)C - mu * mu;
  const float inv = rsqrtf(var + 1e-5f);
  _Float16* orow = out + (size_t)row * C;
  for (int i = tid; i < C; i += 256)
    orow[i] = (_Float16)((xr[i] - mu) * inv * w[i] + b[i]);
}

// ---- fp32 [K][N] -> f16 transposed [N][K] ----
__global__ void wt_f16_kernel(const float* __restrict__ W, _Float16* __restrict__ WT,
                              int K, int N) {
  size_t idx = (size_t)blockIdx.x * blockDim.x + threadIdx.x;
  if (idx >= (size_t)K * N) return;
  const int k = (int)(idx / N), n = (int)(idx % N);
  WT[(size_t)n * K + k] = (_Float16)W[idx];
}

__device__ __forceinline__ void store_tile(const CFrag& c, _Float16* __restrict__ out,
                                           int M, int N, int m0, int n0,
                                           const float* __restrict__ bias,
                                           int lane, int transposed) {
  const int hi = lane >> 4;
  const int n  = n0 + (lane & 15);
  const float bv = bias[n];
  if (!transposed) {
#pragma unroll
    for (int r = 0; r < 8; ++r) {
      const int m = m0 + r + hi * 8;
      out[(size_t)m * N + n] = (_Float16)(c.e[r] + bv);
    }
  } else {
#pragma unroll
    for (int r = 0; r < 8; ++r) {
      const int m = m0 + r + hi * 8;
      out[(size_t)n * M + m] = (_Float16)(c.e[r] + bv);
    }
  }
}

// ---- GEMM: out[m][n] = sum_k A[m][k] * WT[n][k] + bias[n].
// 2x2 register tiling: one wave computes a 32x32 output block (4 WMMA / k-step,
// 2 A-frags + 2 B-frags loaded -> 1KB loaded per 16Kflop WMMA).
__global__ __launch_bounds__(128)
void gemm32_nt_kernel(const _Float16* __restrict__ A, const _Float16* __restrict__ WT,
                      const float* __restrict__ bias, _Float16* __restrict__ out,
                      int M, int N, int K, int store_transposed) {
  const int lane = threadIdx.x & 31;
  const int wid  = threadIdx.x >> 5;
  const int ntiles = N >> 5;                       // 32-wide super-tiles
  const int tile = blockIdx.x * 4 + wid;
  if (tile >= (M >> 5) * ntiles) return;
  const int mt = tile / ntiles, nt = tile % ntiles;
  const _Float16* ap0 = A  + (size_t)(mt * 32) * K;
  const _Float16* ap1 = ap0 + (size_t)16 * K;
  const _Float16* bp0 = WT + (size_t)(nt * 32) * K;
  const _Float16* bp1 = bp0 + (size_t)16 * K;

  CFrag c00, c01, c10, c11;
#pragma unroll
  for (int i = 0; i < 8; ++i) { c00.e[i] = 0.f; c01.e[i] = 0.f; c10.e[i] = 0.f; c11.e[i] = 0.f; }

  for (int k0 = 0; k0 < K; k0 += 32) {
    const v16h a0 = load_a_frag(ap0 + k0, K, lane);
    const v16h a1 = load_a_frag(ap1 + k0, K, lane);
    const v16h b0 = load_b_frag(bp0 + k0, K, lane);
    const v16h b1 = load_b_frag(bp1 + k0, K, lane);
    c00.v = __builtin_amdgcn_wmma_f32_16x16x32_f16(false, a0, false, b0, (short)0, c00.v, false, false);
    c01.v = __builtin_amdgcn_wmma_f32_16x16x32_f16(false, a0, false, b1, (short)0, c01.v, false, false);
    c10.v = __builtin_amdgcn_wmma_f32_16x16x32_f16(false, a1, false, b0, (short)0, c10.v, false, false);
    c11.v = __builtin_amdgcn_wmma_f32_16x16x32_f16(false, a1, false, b1, (short)0, c11.v, false, false);
  }

  const int m0 = mt * 32, n0 = nt * 32;
  store_tile(c00, out, M, N, m0,      n0,      bias, lane, store_transposed);
  store_tile(c01, out, M, N, m0,      n0 + 16, bias, lane, store_transposed);
  store_tile(c10, out, M, N, m0 + 16, n0,      bias, lane, store_transposed);
  store_tile(c11, out, M, N, m0 + 16, n0 + 16, bias, lane, store_transposed);
}

// ---- Fused attention: one block (4 waves) per (b, h, 16-row T tile) ----
// q:    f16 [B*T][D] (head slice contiguous)
// kmat: f16 [B*N][D]
// vT:   f16 [D][B*N]  (d-major so PV B-fragments are contiguous)
__global__ __launch_bounds__(128)
void attn_kernel(const _Float16* __restrict__ q, const _Float16* __restrict__ kmat,
                 const _Float16* __restrict__ vT, float* __restrict__ out) {
  __shared__ float s[16 * NDIM];          // 32 KB: 16 x 512 score strip
  __shared__ float part[4 * 16 * DHEAD];  // 16 KB: per-wave PV partials
  __shared__ float red[16 * 8];
  __shared__ float rowmax[16];
  __shared__ float rowinv[16];

  const int tid  = threadIdx.x;
  const int lane = tid & 31;
  const int wid  = tid >> 5;
  const int m16  = lane & 15;
  const int hi   = lane >> 4;

  int idx = blockIdx.x;
  const int tt = idx & 127; idx >>= 7;   // T/16 = 128 tiles
  const int h  = idx & 15;  idx >>= 4;   // H = 16
  const int b  = idx;                    // B = 4
  const int t0 = tt * 16;

  // ---- S = (Q Khᵀ) * 1/sqrt(64); each wave covers 128 of the 512 columns ----
  const _Float16* qbase = q + ((size_t)(b * TDIM + t0)) * DDIM + h * DHEAD;
  const v16h aq0 = load_a_frag(qbase,      DDIM, lane);   // k = 0..31
  const v16h aq1 = load_a_frag(qbase + 32, DDIM, lane);   // k = 32..63

  for (int ct = 0; ct < 8; ++ct) {
    const int n0 = wid * 128 + ct * 16;
    const _Float16* kb = kmat + ((size_t)(b * NDIM + n0)) * DDIM + h * DHEAD;
    const v16h b0 = load_b_frag(kb,      DDIM, lane);
    const v16h b1 = load_b_frag(kb + 32, DDIM, lane);
    CFrag c;
#pragma unroll
    for (int i = 0; i < 8; ++i) c.e[i] = 0.f;
    c.v = __builtin_amdgcn_wmma_f32_16x16x32_f16(false, aq0, false, b0, (short)0, c.v, false, false);
    c.v = __builtin_amdgcn_wmma_f32_16x16x32_f16(false, aq1, false, b1, (short)0, c.v, false, false);
    const int n = n0 + m16;
#pragma unroll
    for (int r = 0; r < 8; ++r)
      s[(r + hi * 8) * NDIM + n] = c.e[r] * 0.125f;
  }
  __syncthreads();

  // ---- softmax over n (two-pass; keep rows unnormalized, save 1/rowsum) ----
  {
    const int row = tid >> 3, seg = tid & 7;
    const int base = row * NDIM + seg * 64;
    float mx = -3.0e38f;
    for (int j = 0; j < 64; ++j) mx = fmaxf(mx, s[base + j]);
    red[row * 8 + seg] = mx;
  }
  __syncthreads();
  if (tid < 16) {
    float mx = red[tid * 8];
    for (int j = 1; j < 8; ++j) mx = fmaxf(mx, red[tid * 8 + j]);
    rowmax[tid] = mx;
  }
  __syncthreads();
  {
    const int row = tid >> 3, seg = tid & 7;
    const int base = row * NDIM + seg * 64;
    const float mx = rowmax[row];
    float sum = 0.f;
    for (int j = 0; j < 64; ++j) {
      const float e = __expf(s[base + j] - mx);
      s[base + j] = e;
      sum += e;
    }
    red[row * 8 + seg] = sum;
  }
  __syncthreads();
  if (tid < 16) {
    float sum = 0.f;
    for (int j = 0; j < 8; ++j) sum += red[tid * 8 + j];
    rowinv[tid] = 1.0f / sum;
  }
  __syncthreads();

  // ---- Y = P * Vh : each wave contracts its own 128-wide n-range ----
  CFrag acc[4];
#pragma unroll
  for (int dt = 0; dt < 4; ++dt)
#pragma unroll
    for (int i = 0; i < 8; ++i) acc[dt].e[i] = 0.f;

  for (int kc = 0; kc < 4; ++kc) {
    const int k0 = wid * 128 + kc * 32;
    AFrag a;
#pragma unroll
    for (int hh = 0; hh < 16; ++hh) {
      const int col = k0 + hh + hi * 16;
      a.e[hh] = (_Float16)s[m16 * NDIM + col];
    }
#pragma unroll
    for (int dt = 0; dt < 4; ++dt) {
      const _Float16* vp = vT + ((size_t)(h * DHEAD + dt * 16 + m16)) * (BB * NDIM)
                              + b * NDIM + k0 + hi * 16;
      const v16h bf = *(const v16h*)vp;
      acc[dt].v = __builtin_amdgcn_wmma_f32_16x16x32_f16(false, a.v, false, bf, (short)0, acc[dt].v, false, false);
    }
  }

#pragma unroll
  for (int dt = 0; dt < 4; ++dt)
#pragma unroll
    for (int r = 0; r < 8; ++r)
      part[(wid * 16 + r + hi * 8) * DHEAD + dt * 16 + m16] = acc[dt].e[r];
  __syncthreads();

  // ---- reduce the 4 wave partials, apply 1/rowsum, write fp32 output ----
  for (int e = tid * 8; e < tid * 8 + 8; ++e) {
    const int m = e >> 6, d = e & 63;
    const float v = part[(0 * 16 + m) * DHEAD + d] + part[(1 * 16 + m) * DHEAD + d]
                  + part[(2 * 16 + m) * DHEAD + d] + part[(3 * 16 + m) * DHEAD + d];
    out[((size_t)(b * TDIM + t0 + m)) * DDIM + h * DHEAD + d] = v * rowinv[m];
  }
}

extern "C" void kernel_launch(void* const* d_in, const int* in_sizes, int n_in,
                              void* d_out, int out_size, void* d_ws, size_t ws_size,
                              hipStream_t stream) {
  (void)in_sizes; (void)n_in; (void)out_size; (void)ws_size;
  const float* x  = (const float*)d_in[0];
  const float* c  = (const float*)d_in[1];
  const float* nw = (const float*)d_in[2];
  const float* nb = (const float*)d_in[3];
  const float* cw = (const float*)d_in[4];
  const float* cb = (const float*)d_in[5];
  const float* Wq = (const float*)d_in[6];
  const float* bq = (const float*)d_in[7];
  const float* Wk = (const float*)d_in[8];
  const float* bk = (const float*)d_in[9];
  const float* Wv = (const float*)d_in[10];
  const float* bv = (const float*)d_in[11];
  float* out = (float*)d_out;

  _Float16* p = (_Float16*)d_ws;
  _Float16* xn  = p; p += (size_t)BB * TDIM * DDIM;   // 16 MB
  _Float16* cn  = p; p += (size_t)BB * NDIM * LDIM;   //  2 MB
  _Float16* WqT = p; p += (size_t)DDIM * DDIM;        //  2 MB  [D][D]
  _Float16* WkT = p; p += (size_t)DDIM * LDIM;        //  1 MB  [D][L]
  _Float16* WvT = p; p += (size_t)DDIM * LDIM;        //  1 MB  [D][L]
  _Float16* qm  = p; p += (size_t)BB * TDIM * DDIM;   // 16 MB  [B*T][D]
  _Float16* km  = p; p += (size_t)BB * NDIM * DDIM;   //  4 MB  [B*N][D]
  _Float16* vT  = p; p += (size_t)DDIM * BB * NDIM;   //  4 MB  [D][B*N]

  // LayerNorms
  ln_f16_kernel<<<BB * TDIM, 256, 0, stream>>>(x, nw, nb, xn, DDIM);
  ln_f16_kernel<<<BB * NDIM, 256, 0, stream>>>(c, cw, cb, cn, LDIM);

  // Weight convert + transpose to [N][K] f16
  wt_f16_kernel<<<(DDIM * DDIM) / 256, 256, 0, stream>>>(Wq, WqT, DDIM, DDIM);
  wt_f16_kernel<<<(LDIM * DDIM) / 256, 256, 0, stream>>>(Wk, WkT, LDIM, DDIM);
  wt_f16_kernel<<<(LDIM * DDIM) / 256, 256, 0, stream>>>(Wv, WvT, LDIM, DDIM);

  // Projections (WMMA GEMMs, 32x32 per wave): Q, K row-major; V transposed [D][B*N]
  gemm32_nt_kernel<<<((BB * TDIM / 32) * (DDIM / 32)) / 4, 128, 0, stream>>>(
      xn, WqT, bq, qm, BB * TDIM, DDIM, DDIM, 0);
  gemm32_nt_kernel<<<((BB * NDIM / 32) * (DDIM / 32)) / 4, 128, 0, stream>>>(
      cn, WkT, bk, km, BB * NDIM, DDIM, LDIM, 0);
  gemm32_nt_kernel<<<((BB * NDIM / 32) * (DDIM / 32)) / 4, 128, 0, stream>>>(
      cn, WvT, bv, vT, BB * NDIM, DDIM, LDIM, 1);

  // Fused attention: one block per (b, h, 16-row T tile)
  attn_kernel<<<BB * HH * (TDIM / 16), 128, 0, stream>>>(qm, km, vT, out);
}
